// EnhancingFeatureModule_46875273069206
// MI455X (gfx1250) — matrix operand order repaired and checked by reading, not compiled, not run on hardware
//
#include <hip/hip_runtime.h>

typedef __attribute__((ext_vector_type(16))) _Float16 v16h;
typedef __attribute__((ext_vector_type(8)))  _Float16 v8h;
typedef __attribute__((ext_vector_type(8)))  float    v8f;

#define Bsz 8
#define Npt 4096
#define Cch 9
#define Kn  16
#define EPSW 1e-5f

// ---------------- workspace layout (float offsets) ----------------
constexpr size_t OFF_H1    = 0;                                   // f32 [8][64][4096]
constexpr size_t OFF_H2    = OFF_H1 + (size_t)Bsz * 64 * Npt;     // f32 [8][128][4096]
constexpr size_t OFF_XT1   = OFF_H2 + (size_t)Bsz * 128 * Npt;    // f16 [8][4096][64]  (1048576 f32 slots)
constexpr size_t OFF_XT2   = OFF_XT1 + (size_t)Bsz * Npt * 64 / 2;// f16 [8][4096][128] (2097152 f32 slots)
constexpr size_t OFF_WH2   = OFF_XT2 + (size_t)Bsz * Npt * 128 / 2; // f16 128*64  (4096 slots)
constexpr size_t OFF_WH3   = OFF_WH2 + 4096;                      // f16 1024*128 (65536 slots)
constexpr size_t OFF_IDX   = OFF_WH3 + 65536;                     // int [8][4096][16]
constexpr size_t OFF_S1    = OFF_IDX + (size_t)Bsz * Npt * Kn;    // 128
constexpr size_t OFF_S2    = OFF_S1 + 128;                        // 256
constexpr size_t OFF_S3    = OFF_S2 + 256;                        // 2048
constexpr size_t OFF_ES1   = OFF_S3 + 2048;                       // 64
constexpr size_t OFF_ES2   = OFF_ES1 + 64;                        // 32
constexpr size_t OFF_G     = OFF_ES2 + 32;                        // 8192 (atomicMax target)
constexpr size_t OFF_SC1   = OFF_G + 8192;
constexpr size_t OFF_SH1   = OFF_SC1 + 64;
constexpr size_t OFF_SC2   = OFF_SH1 + 64;
constexpr size_t OFF_SH2   = OFF_SC2 + 128;
constexpr size_t OFF_SC3   = OFF_SH2 + 128;
constexpr size_t OFF_SH3   = OFF_SC3 + 1024;
constexpr size_t OFF_ESC1  = OFF_SH3 + 1024;
constexpr size_t OFF_ESH1  = OFF_ESC1 + 32;
constexpr size_t OFF_ESC2  = OFF_ESH1 + 32;
constexpr size_t OFF_ESH2  = OFF_ESC2 + 16;
constexpr size_t WS_FLOATS = OFF_ESH2 + 16;
constexpr size_t ZERO_OFF   = OFF_S1;
constexpr size_t ZERO_COUNT = (OFF_G + 8192) - OFF_S1;

// output layout
constexpr size_t OUT_XYZ   = 0;                                   // [8][3][4096]
constexpr size_t OUT_ENH   = (size_t)Bsz * 3 * Npt;               // [8][25][4096]
constexpr size_t OUT_TRANS = OUT_ENH + (size_t)Bsz * 25 * Npt;    // [8][9]

// ---------------- copy xyz out + enhanced channels 0..8 ----------------
__global__ void copy_kernel(const float* __restrict__ x, float* __restrict__ out) {
  int i = blockIdx.x * 256 + threadIdx.x;
  if (i >= Bsz * Cch * Npt) return;
  int n  = i & (Npt - 1);
  int bc = i >> 12;
  int c  = bc % Cch;
  int b  = bc / Cch;
  float v = x[i];
  out[OUT_ENH + ((size_t)(b * 25 + c)) * Npt + n] = v;
  if (c < 3) out[OUT_XYZ + ((size_t)(b * 3 + c)) * Npt + n] = v;
}

// ---------------- T-net layer 1: 3 -> 64 (scalar) + stats ----------------
__global__ void tnet_l1_kernel(const float* __restrict__ x,
                               const float* __restrict__ w, const float* __restrict__ bias,
                               float* __restrict__ h1, float* __restrict__ stats) {
  int b = blockIdx.x >> 6;
  int c = blockIdx.x & 63;
  int tid = threadIdx.x;
  float w0 = w[c * 3 + 0], w1 = w[c * 3 + 1], w2 = w[c * 3 + 2], bb = bias[c];
  const float* xb = x + (size_t)b * Cch * Npt;
  float* hc = h1 + ((size_t)b * 64 + c) * Npt;
  float s = 0.f, q = 0.f;
  for (int n = tid; n < Npt; n += 256) {
    float y = w0 * xb[n] + w1 * xb[Npt + n] + w2 * xb[2 * Npt + n] + bb;
    hc[n] = y;
    s += y; q += y * y;
  }
  __shared__ float rs[256], rq[256];
  rs[tid] = s; rq[tid] = q;
  __syncthreads();
  for (int off = 128; off > 0; off >>= 1) {
    if (tid < off) { rs[tid] += rs[tid + off]; rq[tid] += rq[tid + off]; }
    __syncthreads();
  }
  if (tid == 0) { atomicAdd(&stats[c], rs[0]); atomicAdd(&stats[64 + c], rq[0]); }
}

// ---------------- finalize BN: stats -> (scale, shift) ----------------
__global__ void finalize_bn_kernel(const float* __restrict__ stats,
                                   const float* __restrict__ gamma, const float* __restrict__ beta,
                                   float* __restrict__ scale, float* __restrict__ shift,
                                   int M, float invCount) {
  int c = blockIdx.x * blockDim.x + threadIdx.x;
  if (c >= M) return;
  float m  = stats[c] * invCount;
  float v  = stats[M + c] * invCount - m * m;
  float rs = rsqrtf(v + EPSW);
  float sc = gamma[c] * rs;
  scale[c] = sc;
  shift[c] = beta[c] - m * sc;
}

// ---------------- BN+ReLU apply, f32 [b][C][n] -> f16 transposed [b][n][C] ----------------
__global__ void apply_bn_f16t_kernel(const float* __restrict__ H,
                                     const float* __restrict__ scale, const float* __restrict__ shift,
                                     _Float16* __restrict__ Xt, int C) {
  int i = blockIdx.x * 256 + threadIdx.x;
  if (i >= Bsz * C * Npt) return;
  int n = i & (Npt - 1);
  int c = (i >> 12) % C;
  int b = (i >> 12) / C;
  float a = fmaxf(0.f, H[i] * scale[c] + shift[c]);
  Xt[((size_t)b * Npt + n) * C + c] = (_Float16)a;
}

// ---------------- weight f32 -> f16 ----------------
__global__ void cvt_w_kernel(const float* __restrict__ W, _Float16* __restrict__ Wh, int sz) {
  int i = blockIdx.x * 256 + threadIdx.x;
  if (i < sz) Wh[i] = (_Float16)W[i];
}

// ---------------- WMMA GEMM: 16x64 tile per wave, f16 inputs ----------------
// Y[b][m][n] = sum_k Wh[m][k] * Xt[b][n][k] + bias[m]
// OUTMODE 0: store Y + stats; 1: stats only; 2: out-BN + ReLU + global max pool.
template <int OUTMODE>
__global__ void wmma_gemm_kernel(const _Float16* __restrict__ Wh, const float* __restrict__ bias,
                                 const _Float16* __restrict__ Xt,
                                 float* __restrict__ Y, float* __restrict__ stats,
                                 const float* __restrict__ outScale, const float* __restrict__ outShift,
                                 float* __restrict__ gmax, int M, int Kd) {
  const int lane = threadIdx.x & 31;
  const int wave = (blockIdx.x * blockDim.x + threadIdx.x) >> 5;
  const int h    = lane >> 4;
  const int l16  = lane & 15;
  const int tilesM = M >> 4;
  const int tm = wave % tilesM;
  const int tp = wave / tilesM;
  if (tp >= (Bsz * Npt) / 64) return;
  const int m0 = tm << 4;
  const int p0 = tp * 64;
  const int b  = p0 >> 12;
  const int n0 = p0 & (Npt - 1);
  const _Float16* Wrow = Wh + (size_t)(m0 + l16) * Kd;
  const _Float16* Xb   = Xt + (size_t)b * Npt * Kd;

  v8f acc[4] = {{}, {}, {}, {}};
  for (int k0 = 0; k0 < Kd; k0 += 32) {
    // A fragment: lane = row m0+l16; elements 0..7 -> K = k0+8h.., 8..15 -> K = k0+16+8h..
    const v8h* pa = (const v8h*)(Wrow + k0);
    v8h alo = pa[h];
    v8h ahi = pa[2 + h];
    v16h afrag;
#pragma unroll
    for (int e = 0; e < 8; ++e) { afrag[e] = alo[e]; afrag[8 + e] = ahi[e]; }
#pragma unroll
    for (int t = 0; t < 4; ++t) {
      // B fragment: lane = col n0+16t+l16; elements 0..7 -> K = k0+8h.., 8..15 -> K = k0+16+8h..
      const v8h* pb = (const v8h*)(Xb + (size_t)(n0 + 16 * t + l16) * Kd + k0);
      v8h blo = pb[h];
      v8h bhi = pb[2 + h];
      v16h bfrag;
#pragma unroll
      for (int e = 0; e < 8; ++e) { bfrag[e] = blo[e]; bfrag[8 + e] = bhi[e]; }
      acc[t] = __builtin_amdgcn_wmma_f32_16x16x32_f16(false, afrag, false, bfrag,
                                                      (short)0, acc[t], false, false);
    }
  }

#pragma unroll
  for (int r = 0; r < 8; ++r) {
    int m = m0 + r + (h << 3);           // D layout: VGPR r -> M = r + 8h, lane -> N
    float bm = bias[m];
    float y0 = acc[0][r] + bm, y1 = acc[1][r] + bm, y2 = acc[2][r] + bm, y3 = acc[3][r] + bm;
    if (OUTMODE == 0) {
      float* yp = Y + ((size_t)b * M + m) * Npt + n0 + l16;
      yp[0] = y0; yp[16] = y1; yp[32] = y2; yp[48] = y3;
    }
    if (OUTMODE <= 1) {
      float s = y0 + y1 + y2 + y3;
      float q = y0 * y0 + y1 * y1 + y2 * y2 + y3 * y3;
      s += __shfl_xor(s, 1, 32); q += __shfl_xor(q, 1, 32);
      s += __shfl_xor(s, 2, 32); q += __shfl_xor(q, 2, 32);
      s += __shfl_xor(s, 4, 32); q += __shfl_xor(q, 4, 32);
      s += __shfl_xor(s, 8, 32); q += __shfl_xor(q, 8, 32);
      if (l16 == 0) { atomicAdd(&stats[m], s); atomicAdd(&stats[M + m], q); }
    }
    if (OUTMODE == 2) {
      float sc = outScale[m], sh = outShift[m];
      float a = fmaxf(0.f, y0 * sc + sh);
      a = fmaxf(a, fmaxf(0.f, y1 * sc + sh));
      a = fmaxf(a, fmaxf(0.f, y2 * sc + sh));
      a = fmaxf(a, fmaxf(0.f, y3 * sc + sh));
      a = fmaxf(a, __shfl_xor(a, 1, 32));
      a = fmaxf(a, __shfl_xor(a, 2, 32));
      a = fmaxf(a, __shfl_xor(a, 4, 32));
      a = fmaxf(a, __shfl_xor(a, 8, 32));
      if (l16 == 0)
        atomicMax((unsigned int*)&gmax[(size_t)b * M + m], __float_as_uint(a)); // post-ReLU >= 0
    }
  }
}

// ---------------- FC head: g(8x1024) -> 512 -> 256 -> 9 (+identity), single block ----
__global__ void fc_kernel(const float* __restrict__ g,
                          const float* __restrict__ fw1, const float* __restrict__ fb1,
                          const float* __restrict__ fg4, const float* __restrict__ fbe4,
                          const float* __restrict__ fw2, const float* __restrict__ fb2,
                          const float* __restrict__ fg5, const float* __restrict__ fbe5,
                          const float* __restrict__ fw3, const float* __restrict__ fb3,
                          float* __restrict__ trans_out) {
  __shared__ float s1[8 * 512];
  __shared__ float s2[8 * 256];
  int tid = threadIdx.x;   // 256
  for (int o = tid; o < 8 * 512; o += 256) {
    int s = o >> 9, j = o & 511;
    const float* gs = g + (size_t)s * 1024;
    const float* w  = fw1 + (size_t)j * 1024;
    float acc = fb1[j];
    for (int c = 0; c < 1024; ++c) acc += gs[c] * w[c];
    s1[o] = acc;
  }
  __syncthreads();
  for (int j = tid; j < 512; j += 256) {
    float m = 0.f;
    for (int s = 0; s < 8; ++s) m += s1[s * 512 + j];
    m *= 0.125f;
    float v = 0.f;
    for (int s = 0; s < 8; ++s) { float d = s1[s * 512 + j] - m; v += d * d; }
    v *= 0.125f;
    float rs = rsqrtf(v + EPSW), sc = fg4[j] * rs, sh = fbe4[j] - m * sc;
    for (int s = 0; s < 8; ++s) s1[s * 512 + j] = fmaxf(0.f, s1[s * 512 + j] * sc + sh);
  }
  __syncthreads();
  for (int o = tid; o < 8 * 256; o += 256) {
    int s = o >> 8, j = o & 255;
    const float* w = fw2 + (size_t)j * 512;
    float acc = fb2[j];
    for (int c = 0; c < 512; ++c) acc += s1[s * 512 + c] * w[c];
    s2[o] = acc;
  }
  __syncthreads();
  if (tid < 256) {
    int j = tid;
    float m = 0.f;
    for (int s = 0; s < 8; ++s) m += s2[s * 256 + j];
    m *= 0.125f;
    float v = 0.f;
    for (int s = 0; s < 8; ++s) { float d = s2[s * 256 + j] - m; v += d * d; }
    v *= 0.125f;
    float rs = rsqrtf(v + EPSW), sc = fg5[j] * rs, sh = fbe5[j] - m * sc;
    for (int s = 0; s < 8; ++s) s2[s * 256 + j] = fmaxf(0.f, s2[s * 256 + j] * sc + sh);
  }
  __syncthreads();
  if (tid < 72) {
    int s = tid / 9, j = tid % 9;
    const float* w = fw3 + (size_t)j * 256;
    float acc = fb3[j] + ((j == 0 || j == 4 || j == 8) ? 1.f : 0.f);
    for (int c = 0; c < 256; ++c) acc += s2[s * 256 + c] * w[c];
    trans_out[s * 9 + j] = acc;
  }
}

// ---------------- KNN: per point top-16 nearest (incl. self) ----------------
__global__ void knn_kernel(const float* __restrict__ x, int* __restrict__ idxbuf) {
  int tid = threadIdx.x;
  int b = blockIdx.x >> 4;
  int n = ((blockIdx.x & 15) << 8) + tid;
  const float* xb = x + (size_t)b * Cch * Npt;
  float px = xb[n], py = xb[Npt + n], pz = xb[2 * Npt + n];
  float kd[Kn]; int ki[Kn];
#pragma unroll
  for (int t = 0; t < Kn; ++t) { kd[t] = 3.4e38f; ki[t] = 0; }
  __shared__ float tile[3][128];
  for (int base = 0; base < Npt; base += 128) {
    for (int i = tid; i < 3 * 128; i += 256) {
      int ch = i >> 7, jj = i & 127;
      tile[ch][jj] = xb[ch * Npt + base + jj];
    }
    __syncthreads();
    for (int jj = 0; jj < 128; ++jj) {
      float dx = px - tile[0][jj], dy = py - tile[1][jj], dz = pz - tile[2][jj];
      float d = dx * dx + dy * dy + dz * dz;
      int j = base + jj;
      if (d < kd[Kn - 1]) {
#pragma unroll
        for (int t = Kn - 1; t > 0; --t) {
          if (kd[t] > d) {
            bool up = kd[t - 1] > d;
            kd[t] = up ? kd[t - 1] : d;
            ki[t] = up ? ki[t - 1] : j;
          }
        }
        if (kd[0] > d) { kd[0] = d; ki[0] = j; }
      }
    }
    __syncthreads();
  }
  int* ip = idxbuf + ((size_t)b * Npt + n) * Kn;
#pragma unroll
  for (int t = 0; t < Kn; ++t) ip[t] = ki[t];
}

// ---------------- Edge MLP passes ----------------
// MODE 0: stats of e1 (32ch). MODE 1: stats of e2 (16ch). MODE 2: max_k -> enhanced.
template <int MODE>
__global__ void edge_kernel(const float* __restrict__ x, const int* __restrict__ idxbuf,
                            const float* __restrict__ ew1, const float* __restrict__ eb1,
                            const float* __restrict__ esc1, const float* __restrict__ esh1,
                            const float* __restrict__ ew2, const float* __restrict__ eb2,
                            const float* __restrict__ esc2, const float* __restrict__ esh2,
                            float* __restrict__ dst) {
  __shared__ float sw1[32 * 21], sb1[32];
  __shared__ float ssc1[32], ssh1[32];
  __shared__ float sw2[16 * 32], sb2[16], ssc2[16], ssh2[16];
  __shared__ float rsum[32], rsq[32];
  int tid = threadIdx.x;
  for (int i = tid; i < 32 * 21; i += 256) sw1[i] = ew1[i];
  if (tid < 32) {
    sb1[tid] = eb1[tid];
    rsum[tid] = 0.f; rsq[tid] = 0.f;
    if (MODE >= 1) { ssc1[tid] = esc1[tid]; ssh1[tid] = esh1[tid]; }
  }
  if (MODE >= 1) {
    for (int i = tid; i < 16 * 32; i += 256) sw2[i] = ew2[i];
    if (tid < 16) {
      sb2[tid] = eb2[tid];
      if (MODE == 2) { ssc2[tid] = esc2[tid]; ssh2[tid] = esh2[tid]; }
    }
  }
  __syncthreads();

  int p = blockIdx.x * 256 + tid;
  int b = p >> 12, n = p & (Npt - 1);
  const float* xb = x + (size_t)b * Cch * Npt;
  float cf[9];
#pragma unroll
  for (int c = 0; c < 9; ++c) cf[c] = xb[c * Npt + n];
  const int* ip = idxbuf + (size_t)p * Kn;

  float acc1[32]; float acc2[32];
  const int NACC = (MODE == 0) ? 32 : 16;
#pragma unroll
  for (int o = 0; o < 32; ++o) { acc1[o] = (MODE == 2) ? -3.4e38f : 0.f; acc2[o] = 0.f; }

  for (int k = 0; k < Kn; ++k) {
    int j = ip[k];
    float ef[21];
#pragma unroll
    for (int c = 0; c < 9; ++c) {
      float nf = xb[c * Npt + j];
      ef[c] = cf[c];
      ef[9 + c] = nf - cf[c];
    }
    ef[18] = ef[9]; ef[19] = ef[10]; ef[20] = ef[11];   // rel_xyz == nfeat-cfeat on xyz ch.
    float a1[32];
#pragma unroll
    for (int o = 0; o < 32; ++o) {
      const float* w = &sw1[o * 21];
      float e = sb1[o];
#pragma unroll
      for (int c = 0; c < 21; ++c) e += w[c] * ef[c];
      if (MODE == 0) { acc1[o] += e; acc2[o] += e * e; }
      else a1[o] = fmaxf(0.f, e * ssc1[o] + ssh1[o]);
    }
    if (MODE >= 1) {
#pragma unroll
      for (int o = 0; o < 16; ++o) {
        const float* w = &sw2[o * 32];
        float e = sb2[o];
#pragma unroll
        for (int c = 0; c < 32; ++c) e += w[c] * a1[c];
        if (MODE == 1) { acc1[o] += e; acc2[o] += e * e; }
        else acc1[o] = fmaxf(acc1[o], fmaxf(0.f, e * ssc2[o] + ssh2[o]));
      }
    }
  }

  if (MODE == 2) {
    float* eb = dst + OUT_ENH + (size_t)b * 25 * Npt;
#pragma unroll
    for (int o = 0; o < 16; ++o) eb[(size_t)(9 + o) * Npt + n] = acc1[o];
  } else {
    for (int o = 0; o < NACC; ++o) { atomicAdd(&rsum[o], acc1[o]); atomicAdd(&rsq[o], acc2[o]); }
    __syncthreads();
    if (tid < NACC) { atomicAdd(&dst[tid], rsum[tid]); atomicAdd(&dst[NACC + tid], rsq[tid]); }
  }
}

extern "C" void kernel_launch(void* const* d_in, const int* in_sizes, int n_in,
                              void* d_out, int out_size, void* d_ws, size_t ws_size,
                              hipStream_t stream) {
  const float* x    = (const float*)d_in[0];
  const float* tw1  = (const float*)d_in[1];
  const float* tb1  = (const float*)d_in[2];
  const float* tg1  = (const float*)d_in[3];
  const float* tbe1 = (const float*)d_in[4];
  const float* tw2  = (const float*)d_in[5];
  const float* tb2  = (const float*)d_in[6];
  const float* tg2  = (const float*)d_in[7];
  const float* tbe2 = (const float*)d_in[8];
  const float* tw3  = (const float*)d_in[9];
  const float* tb3  = (const float*)d_in[10];
  const float* tg3  = (const float*)d_in[11];
  const float* tbe3 = (const float*)d_in[12];
  const float* fw1  = (const float*)d_in[13];
  const float* fb1  = (const float*)d_in[14];
  const float* fg4  = (const float*)d_in[15];
  const float* fbe4 = (const float*)d_in[16];
  const float* fw2  = (const float*)d_in[17];
  const float* fb2  = (const float*)d_in[18];
  const float* fg5  = (const float*)d_in[19];
  const float* fbe5 = (const float*)d_in[20];
  const float* fw3  = (const float*)d_in[21];
  const float* fb3  = (const float*)d_in[22];
  const float* ew1  = (const float*)d_in[23];
  const float* eb1  = (const float*)d_in[24];
  const float* eg1  = (const float*)d_in[25];
  const float* ebe1 = (const float*)d_in[26];
  const float* ew2  = (const float*)d_in[27];
  const float* eb2  = (const float*)d_in[28];
  const float* eg2  = (const float*)d_in[29];
  const float* ebe2 = (const float*)d_in[30];

  float* out = (float*)d_out;
  float* ws  = (float*)d_ws;
  if (ws_size < WS_FLOATS * sizeof(float)) return;

  float*     h1   = ws + OFF_H1;
  float*     h2   = ws + OFF_H2;
  _Float16*  xt1  = (_Float16*)(ws + OFF_XT1);
  _Float16*  xt2  = (_Float16*)(ws + OFF_XT2);
  _Float16*  wh2  = (_Float16*)(ws + OFF_WH2);
  _Float16*  wh3  = (_Float16*)(ws + OFF_WH3);
  int*       idxb = (int*)(ws + OFF_IDX);
  float* s1   = ws + OFF_S1;
  float* s2   = ws + OFF_S2;
  float* s3   = ws + OFF_S3;
  float* es1  = ws + OFF_ES1;
  float* es2  = ws + OFF_ES2;
  float* g    = ws + OFF_G;
  float* sc1  = ws + OFF_SC1;  float* sh1  = ws + OFF_SH1;
  float* sc2  = ws + OFF_SC2;  float* sh2  = ws + OFF_SH2;
  float* sc3  = ws + OFF_SC3;  float* sh3  = ws + OFF_SH3;
  float* esc1 = ws + OFF_ESC1; float* esh1 = ws + OFF_ESH1;
  float* esc2 = ws + OFF_ESC2; float* esh2 = ws + OFF_ESH2;

  hipMemsetAsync(ws + ZERO_OFF, 0, ZERO_COUNT * sizeof(float), stream);

  // passthrough outputs
  copy_kernel<<<(Bsz * Cch * Npt + 255) / 256, 256, 0, stream>>>(x, out);

  // weight conversions (no deps)
  cvt_w_kernel<<<(128 * 64 + 255) / 256, 256, 0, stream>>>(tw2, wh2, 128 * 64);
  cvt_w_kernel<<<(1024 * 128 + 255) / 256, 256, 0, stream>>>(tw3, wh3, 1024 * 128);

  const float invBN  = 1.f / (float)(Bsz * Npt);
  const float invBNK = 1.f / (float)(Bsz * Npt * Kn);

  // T-net layer 1 (3->64)
  tnet_l1_kernel<<<Bsz * 64, 256, 0, stream>>>(x, tw1, tb1, h1, s1);
  finalize_bn_kernel<<<1, 64, 0, stream>>>(s1, tg1, tbe1, sc1, sh1, 64, invBN);
  apply_bn_f16t_kernel<<<(Bsz * 64 * Npt + 255) / 256, 256, 0, stream>>>(h1, sc1, sh1, xt1, 64);

  // layer 2 (64->128) WMMA: store pre-act h2 + stats
  {
    int waves = (128 / 16) * ((Bsz * Npt) / 64);
    wmma_gemm_kernel<0><<<waves / 4, 128, 0, stream>>>(wh2, tb2, xt1, h2, s2,
                                                       nullptr, nullptr, nullptr, 128, 64);
  }
  finalize_bn_kernel<<<1, 128, 0, stream>>>(s2, tg2, tbe2, sc2, sh2, 128, invBN);
  apply_bn_f16t_kernel<<<(Bsz * 128 * Npt + 255) / 256, 256, 0, stream>>>(h2, sc2, sh2, xt2, 128);

  // layer 3 (128->1024) WMMA pass A: stats only
  int waves3 = (1024 / 16) * ((Bsz * Npt) / 64);
  wmma_gemm_kernel<1><<<waves3 / 4, 128, 0, stream>>>(wh3, tb3, xt2, nullptr, s3,
                                                      nullptr, nullptr, nullptr, 1024, 128);
  finalize_bn_kernel<<<4, 256, 0, stream>>>(s3, tg3, tbe3, sc3, sh3, 1024, invBN);

  // layer 3 pass B: recompute, BN+ReLU, global max pool -> g
  wmma_gemm_kernel<2><<<waves3 / 4, 128, 0, stream>>>(wh3, tb3, xt2, nullptr, nullptr,
                                                      sc3, sh3, g, 1024, 128);

  // FC head -> trans
  fc_kernel<<<1, 256, 0, stream>>>(g, fw1, fb1, fg4, fbe4, fw2, fb2, fg5, fbe5,
                                   fw3, fb3, out + OUT_TRANS);

  // KNN + edge MLP
  knn_kernel<<<Bsz * (Npt / 256), 256, 0, stream>>>(x, idxb);
  edge_kernel<0><<<(Bsz * Npt) / 256, 256, 0, stream>>>(x, idxb, ew1, eb1, nullptr, nullptr,
                                                        ew2, eb2, nullptr, nullptr, es1);
  finalize_bn_kernel<<<1, 32, 0, stream>>>(es1, eg1, ebe1, esc1, esh1, 32, invBNK);
  edge_kernel<1><<<(Bsz * Npt) / 256, 256, 0, stream>>>(x, idxb, ew1, eb1, esc1, esh1,
                                                        ew2, eb2, nullptr, nullptr, es2);
  finalize_bn_kernel<<<1, 16, 0, stream>>>(es2, eg2, ebe2, esc2, esh2, 16, invBNK);
  edge_kernel<2><<<(Bsz * Npt) / 256, 256, 0, stream>>>(x, idxb, ew1, eb1, esc1, esh1,
                                                        ew2, eb2, esc2, esh2, out);
}
